// Encoder_27590869910018
// MI455X (gfx1250) — matrix operand order repaired and checked by reading, not compiled, not run on hardware
//
#include <hip/hip_runtime.h>
#include <hip/hip_bf16.h>

// ---------------- problem constants ----------------
#define Bb   64
#define Ss   336
#define Pp   128
#define Ff   16
#define Hh   1024
#define Mtok (Bb * Ss)        // 21504 tokens, = 336 * 64 exactly

#define KSTEP 64              // K per LDS stage
#define PADK  72              // 64 + 8 bf16 pad -> 144B row stride (16B aligned)

typedef unsigned short u16;
typedef __attribute__((ext_vector_type(16))) __bf16 v16bf;
typedef __attribute__((ext_vector_type(8)))  float  v8f;
typedef __attribute__((ext_vector_type(4)))  unsigned int u32x4;

#define WMMA_BF16(a, b, c) \
  __builtin_amdgcn_wmma_f32_16x16x32_bf16(false, (a), false, (b), (short)0, (c), false, false)

// float -> bf16 round-to-nearest-even
__device__ __forceinline__ u16 f2bf(float f) {
  unsigned u = __float_as_uint(f);
  u += 0x7FFFu + ((u >> 16) & 1u);
  return (u16)(u >> 16);
}

// A fragment (16x32 bf16): lane = M (mod 16); lo lanes hold K {kk+0..7, kk+16..23},
// hi lanes hold K {kk+8..15, kk+24..31}.  row = base + m*ld + kk.
__device__ __forceinline__ v16bf ldA(const u16* __restrict__ row, bool hi) {
  const u16* p = row + (hi ? 8 : 0);
  union { u32x4 q[2]; v16bf v; } t;
  t.q[0] = *(const u32x4*)(p);
  t.q[1] = *(const u32x4*)(p + 16);
  return t.v;
}

// B fragment (32x16 bf16): lane = N (mod 16); lo lanes hold K {kk+0..7, kk+16..23},
// hi lanes hold K {kk+8..15, kk+24..31}.  row = base + n*ld + kk.
__device__ __forceinline__ v16bf ldB(const u16* __restrict__ row, bool hi) {
  const u16* p = row + (hi ? 8 : 0);
  union { u32x4 q[2]; v16bf v; } t;
  t.q[0] = *(const u32x4*)(p);
  t.q[1] = *(const u32x4*)(p + 16);
  return t.v;
}

// CDNA5 async global->LDS copy (16B per lane), tracked by ASYNCcnt.
// LDS address = low 32 bits of the generic pointer.
__device__ __forceinline__ void async_b128(unsigned lds_off, const void* gaddr) {
  asm volatile("global_load_async_to_lds_b128 %0, %1, off"
               :: "v"(lds_off), "v"(gaddr) : "memory");
}
__device__ __forceinline__ void wait_async8() {
  asm volatile("s_wait_asynccnt 0x8" ::: "memory");
}
__device__ __forceinline__ void wait_async0() {
  asm volatile("s_wait_asynccnt 0x0" ::: "memory");
}

// ---------------- f32 -> bf16 weight conversion ----------------
__global__ __launch_bounds__(256) void cvt_f32_bf16(const float* __restrict__ in,
                                                    u16* __restrict__ out, int n) {
  int i = blockIdx.x * 256 + threadIdx.x;
  if (i < n) out[i] = f2bf(in[i]);
}

// ---------------- per-point MLPs (f32 VALU, weights in LDS) ----------------
// grid: (Mtok/256, P), block: 256.  feat bf16 [Mtok, P*8]
__global__ __launch_bounds__(256) void point_mlp(
    const float* __restrict__ x,  const float* __restrict__ W1,
    const float* __restrict__ b1, const float* __restrict__ W2,
    const float* __restrict__ b2, u16* __restrict__ feat) {
  __shared__ float sW1[256], sW2[128], sb1[16], sb2[8];
  const int p   = blockIdx.y;
  const int tid = threadIdx.x;
  sW1[tid] = W1[p * 256 + tid];
  if (tid < 128) sW2[tid] = W2[p * 128 + tid];
  if (tid < 16)  sb1[tid] = b1[p * 16 + tid];
  if (tid < 8)   sb2[tid] = b2[p * 8 + tid];
  __syncthreads();

  const int token = blockIdx.x * 256 + tid;           // Mtok % 256 == 0
  const float* xp = x + ((size_t)token * Pp + p) * Ff;
  float xi[16];
  #pragma unroll
  for (int f = 0; f < 16; ++f) xi[f] = xp[f];

  float h1[16];
  #pragma unroll
  for (int k = 0; k < 16; ++k) {
    float s = sb1[k];
    #pragma unroll
    for (int f = 0; f < 16; ++f) s = fmaf(sW1[k * 16 + f], xi[f], s);
    h1[k] = fmaxf(s, 0.f);
  }
  #pragma unroll
  for (int o = 0; o < 8; ++o) {
    float s = sb2[o];
    #pragma unroll
    for (int k = 0; k < 16; ++k) s = fmaf(sW2[o * 16 + k], h1[k], s);
    feat[(size_t)token * (Pp * 8) + p * 8 + o] = f2bf(fmaxf(s, 0.f));
  }
}

// ---------------- WMMA GEMM with async-LDS double buffering ----------------
// Out[m,n] = act( A[m,:] . W[n,:] + bias[n] ),  A bf16 [M,K], W bf16 [N,K].
// grid: (M/64, N/64), block: 128 (4 waves).  Block tile 64x64, K staged 64 at a
// time into LDS via global_load_async_to_lds_b128 (double buffered), WMMA reads
// fragments from LDS (B tiles shared by all 4 waves -> 4x less global traffic).
template <bool RELU, bool OUT_BF16>
__global__ __launch_bounds__(128) void gemm_bf16(
    const u16* __restrict__ A, const u16* __restrict__ W,
    const float* __restrict__ bias, void* __restrict__ Out, int N, int K) {
  __shared__ __align__(16) u16 sA[2][64 * PADK];
  __shared__ __align__(16) u16 sB[2][64 * PADK];

  const int tid  = threadIdx.x;
  const int lane = tid & 31;
  const int wave = tid >> 5;
  const int l15  = lane & 15;
  const bool hi  = lane >= 16;
  const int gm0  = blockIdx.x * 64;
  const int gn0  = blockIdx.y * 64;

  // 512 x 16B copies per 64x64 bf16 tile; 128 threads -> 4 copies each per matrix.
  // copy c: row r = c>>3, 16B segment s = c&7.
  auto issue = [&](int ks, int buf) {
    const u16* gA = A + (size_t)gm0 * K + ks * KSTEP;
    const u16* gB = W + (size_t)gn0 * K + ks * KSTEP;
    #pragma unroll
    for (int i = 0; i < 4; ++i) {
      const int c = tid + i * 128;
      const int r = c >> 3, s = c & 7;
      async_b128((unsigned)(size_t)&sA[buf][r * PADK + s * 8],
                 gA + (size_t)r * K + s * 8);
      async_b128((unsigned)(size_t)&sB[buf][r * PADK + s * 8],
                 gB + (size_t)r * K + s * 8);
    }
  };

  v8f acc0 = {}, acc1 = {}, acc2 = {}, acc3 = {};
  const int nK = K / KSTEP;

  issue(0, 0);
  for (int ks = 0; ks < nK; ++ks) {
    const int buf = ks & 1;
    if (ks + 1 < nK) { issue(ks + 1, buf ^ 1); wait_async8(); }
    else             { wait_async0(); }
    __syncthreads();   // tile ks visible to all waves

    const u16* tA  = &sA[buf][(wave * 16 + l15) * PADK];
    const u16* tB0 = &sB[buf][( 0 + l15) * PADK];
    const u16* tB1 = &sB[buf][(16 + l15) * PADK];
    const u16* tB2 = &sB[buf][(32 + l15) * PADK];
    const u16* tB3 = &sB[buf][(48 + l15) * PADK];
    #pragma unroll
    for (int kk = 0; kk < KSTEP; kk += 32) {
      v16bf a = ldA(tA + kk, hi);
      acc0 = WMMA_BF16(a, ldB(tB0 + kk, hi), acc0);
      acc1 = WMMA_BF16(a, ldB(tB1 + kk, hi), acc1);
      acc2 = WMMA_BF16(a, ldB(tB2 + kk, hi), acc2);
      acc3 = WMMA_BF16(a, ldB(tB3 + kk, hi), acc3);
    }
    __syncthreads();   // done reading buf before it is overwritten
  }

  const int mrow = gm0 + wave * 16 + (hi ? 8 : 0);
  #pragma unroll
  for (int j = 0; j < 4; ++j) {
    v8f acc = (j == 0) ? acc0 : (j == 1) ? acc1 : (j == 2) ? acc2 : acc3;
    const int n  = gn0 + j * 16 + l15;
    const float bi = bias[n];
    #pragma unroll
    for (int i = 0; i < 8; ++i) {
      float v = acc[i] + bi;
      if (RELU) v = fmaxf(v, 0.f);
      const size_t idx = (size_t)(mrow + i) * N + n;
      if (OUT_BF16) ((u16*)Out)[idx]   = f2bf(v);
      else          ((float*)Out)[idx] = v;
    }
  }
}

// ---------------- one GRU time step ----------------
// grid: H/16 = 64 blocks over columns, block: 128 (4 waves over the 64 batch rows).
// Latency-critical serial chain: keep direct global->WMMA (no barriers), A
// fragment of h shared by the 3 gate WMMAs, gate math fused.
__global__ __launch_bounds__(128) void gru_step(
    const u16* __restrict__ hIn, u16* __restrict__ hOut,
    const u16* __restrict__ Whh, const float* __restrict__ xW,
    const float* __restrict__ bhh, float* __restrict__ out,
    float* __restrict__ hLast, int t) {
  const int lane = threadIdx.x & 31;
  const int wave = threadIdx.x >> 5;
  const int l15  = lane & 15;
  const bool hi  = lane >= 16;
  const int m0   = wave * 16;
  const int n0   = blockIdx.x * 16;

  const u16* Ar = hIn + (size_t)(m0 + l15) * Hh;
  const u16* Br = Whh + (size_t)(0 * Hh + n0 + l15) * Hh;
  const u16* Bz = Whh + (size_t)(1 * Hh + n0 + l15) * Hh;
  const u16* Bn = Whh + (size_t)(2 * Hh + n0 + l15) * Hh;

  v8f ar = {}, az = {}, an = {};
  for (int k = 0; k < Hh; k += 32) {
    v16bf a = ldA(Ar + k, hi);
    ar = WMMA_BF16(a, ldB(Br + k, hi), ar);
    az = WMMA_BF16(a, ldB(Bz + k, hi), az);
    an = WMMA_BF16(a, ldB(Bn + k, hi), an);
  }

  const int col = n0 + l15;
  const float br = bhh[col];
  const float bz = bhh[Hh + col];
  const float bn = bhh[2 * Hh + col];
  const int mbase = m0 + (hi ? 8 : 0);

  #pragma unroll
  for (int i = 0; i < 8; ++i) {
    const int b = mbase + i;
    const size_t xrow = ((size_t)b * Ss + t) * (size_t)(3 * Hh);
    const float xr = xW[xrow + col];
    const float xz = xW[xrow + Hh + col];
    const float xn = xW[xrow + 2 * Hh + col];
    const float r  = 1.f / (1.f + __expf(-(xr + ar[i] + br)));
    const float z  = 1.f / (1.f + __expf(-(xz + az[i] + bz)));
    const float ng = tanhf(xn + r * (an[i] + bn));
    const float hp = (t == 0) ? 0.f : out[((size_t)b * Ss + (t - 1)) * Hh + col];
    const float hn = (1.f - z) * ng + z * hp;
    out[((size_t)b * Ss + t) * Hh + col] = hn;
    hOut[(size_t)b * Hh + col] = f2bf(hn);
    if (hLast) hLast[(size_t)b * Hh + col] = hn;
  }
}

// ---------------- host launcher ----------------
extern "C" void kernel_launch(void* const* d_in, const int* in_sizes, int n_in,
                              void* d_out, int out_size, void* d_ws, size_t ws_size,
                              hipStream_t stream) {
  (void)in_sizes; (void)n_in; (void)out_size; (void)ws_size;

  const float* x   = (const float*)d_in[0];
  const float* W1  = (const float*)d_in[1];
  const float* b1  = (const float*)d_in[2];
  const float* W2  = (const float*)d_in[3];
  const float* b2  = (const float*)d_in[4];
  const float* Wl  = (const float*)d_in[5];
  const float* bl  = (const float*)d_in[6];
  const float* Wih = (const float*)d_in[7];
  const float* Whh = (const float*)d_in[8];
  const float* bih = (const float*)d_in[9];
  const float* bhh = (const float*)d_in[10];

  char*  ws  = (char*)d_ws;
  size_t off = 0;
  auto alloc = [&](size_t bytes) -> void* {
    void* p = ws + off;
    off = (off + bytes + 255) & ~(size_t)255;
    return p;
  };
  u16*   featB = (u16*)  alloc((size_t)Mtok * Hh * 2);        // 44 MB
  u16*   linB  = (u16*)  alloc((size_t)Mtok * Hh * 2);        // 44 MB
  u16*   WlB   = (u16*)  alloc((size_t)Hh * Hh * 2);          //  2 MB
  u16*   WihB  = (u16*)  alloc((size_t)3 * Hh * Hh * 2);      //  6 MB
  u16*   WhhB  = (u16*)  alloc((size_t)3 * Hh * Hh * 2);      //  6 MB
  float* xW    = (float*)alloc((size_t)Mtok * 3 * Hh * 4);    // 264 MB
  u16*   hPing = (u16*)  alloc((size_t)Bb * Hh * 2);
  u16*   hPong = (u16*)  alloc((size_t)Bb * Hh * 2);

  float* out   = (float*)d_out;                 // [B,S,H]
  float* hLast = out + (size_t)Bb * Ss * Hh;    // [1,B,H]

  // 1) weights -> bf16
  cvt_f32_bf16<<<(Hh * Hh) / 256,     256, 0, stream>>>(Wl,  WlB,  Hh * Hh);
  cvt_f32_bf16<<<(3 * Hh * Hh) / 256, 256, 0, stream>>>(Wih, WihB, 3 * Hh * Hh);
  cvt_f32_bf16<<<(3 * Hh * Hh) / 256, 256, 0, stream>>>(Whh, WhhB, 3 * Hh * Hh);

  // 2) per-point MLPs -> feat (bf16)
  dim3 gMlp(Mtok / 256, Pp);
  point_mlp<<<gMlp, 256, 0, stream>>>(x, W1, b1, W2, b2, featB);

  // 3) lin = relu(feat . Wl^T + bl)  (bf16 out)
  dim3 gLin(Mtok / 64, Hh / 64);
  gemm_bf16<true, true><<<gLin, 128, 0, stream>>>(featB, WlB, bl, linB, Hh, Hh);

  // 4) xW = lin . W_ih^T + b_ih  (f32 out, all timesteps)
  dim3 gXW(Mtok / 64, (3 * Hh) / 64);
  gemm_bf16<false, false><<<gXW, 128, 0, stream>>>(linB, WihB, bih, xW, 3 * Hh, Hh);

  // 5) GRU scan (h0 = 0)
  hipMemsetAsync(hPing, 0, (size_t)Bb * Hh * 2, stream);
  for (int t = 0; t < Ss; ++t) {
    const u16* hin = (t & 1) ? hPong : hPing;
    u16*       hou = (t & 1) ? hPing : hPong;
    float* hl = (t == Ss - 1) ? hLast : nullptr;
    gru_step<<<Hh / 16, 128, 0, stream>>>(hin, hou, WhhB, xW, bhh, out, hl, t);
  }
}
